// GNN_1_21002390078195
// MI455X (gfx1250) — compile-verified, matrix operand order checked by reading, not run on hardware
//
#include <hip/hip_runtime.h>
#include <hip/hip_bf16.h>

// ---------------------------------------------------------------------------
// GNN NNConv x3 + pooling + head, fused for MI455X (gfx1250, wave32, WMMA).
//
// msg_e = x_src(e)^T * reshape(h_e@w2 + b2, [mi,mo]) is bilinear, so
// msg = Z @ w2' with Z[e, k*mi+i] = h_e[k]*x_src[e][i] and w2' = w2 viewed as
// [128*mi, mo] row-major. b2 folds in as mi extra K-rows with h == 1.
// All GEMMs are f16 WMMA 16x16x32 with f32 accumulate; weights pre-swizzled
// into per-lane fragment order so each B fragment is 2 contiguous b128 loads.
// Edge-message kernel blocks M by 5 tiles (80 edges) so every B fragment is
// reused by 5 independent WMMA chains (hides WMMA RAW latency, 5x less L2
// weight traffic). A fragments are built with ds_load_b128 + v_pk_mul_f16.
// ---------------------------------------------------------------------------

#define NN_NODES 30000
#define NN_EDGES 60000
#define NN_GRAPHS 1024

typedef _Float16 v16h __attribute__((ext_vector_type(16)));
typedef _Float16 v8h  __attribute__((ext_vector_type(8)));
typedef float    v8f  __attribute__((ext_vector_type(8)));

// K index inside a 16x32 16-bit fragment for lane-half `half`, element j,
// per CDNA5 ISA 7.12.2 16-bit A layout. For j=0..7:  K = 8*half + j
//                                        for j=8..15: K = 16 + 8*half + (j-8)
// => each lane's 16 elements are two contiguous, 16B-aligned runs of 8.
__device__ __forceinline__ int kmap(int j, int half) {
  return j + 8 * half + ((j >= 8) ? 8 : 0);
}

// Load a 16-element A fragment from LDS row `base` (two b128 loads).
__device__ __forceinline__ v16h load_a_frag(const _Float16* base, int half) {
  v8h lo = *(const v8h*)(base + 8 * half);
  v8h hi = *(const v8h*)(base + 16 + 8 * half);
  return __builtin_shufflevector(lo, hi, 0, 1, 2, 3, 4, 5, 6, 7, 8, 9, 10, 11,
                                 12, 13, 14, 15);
}

// ---------------------------------------------------------------------------
// Pre-swizzle a [Kmain(+Ktail), mo] f32 matrix into per-lane f16 B fragments:
// out index (c, t, L, j) <- M[c*32 + kmap(j, L>>4)][t*16 + (L&15)]
// Rows >= Kmain come from Mtail (bias fold); rows beyond that are zero pad.
// ---------------------------------------------------------------------------
__global__ void k_swizzleB(const float* __restrict__ M,
                           const float* __restrict__ Mtail,
                           int Kmain, int Ktail, int mo,
                           _Float16* __restrict__ out, int total) {
  int T = mo >> 4;
  for (int idx = blockIdx.x * blockDim.x + threadIdx.x; idx < total;
       idx += gridDim.x * blockDim.x) {
    int j = idx & 15;
    int L = (idx >> 4) & 31;
    int t = (idx >> 9) % T;
    int c = idx / (512 * T);
    int n = L & 15, half = L >> 4;
    int kk = c * 32 + kmap(j, half);
    int col = t * 16 + n;
    float v = 0.f;
    if (kk < Kmain)              v = M[(size_t)kk * mo + col];
    else if (kk - Kmain < Ktail) v = Mtail[(size_t)(kk - Kmain) * mo + col];
    out[idx] = (_Float16)v;
  }
}

// ---------------------------------------------------------------------------
// h = relu(edge_attr @ w1 + b1) -> f16 [E,128].  16 edges/block, 4 waves,
// each wave does 2 of the 8 N-tiles.  K=16 zero-padded to 32.
// ---------------------------------------------------------------------------
__global__ __launch_bounds__(128) void k_edge_h(
    const float* __restrict__ ea,       // [E,16]
    const _Float16* __restrict__ w1f,   // swizzled frags, C=1, T=8
    const float* __restrict__ b1,       // [128]
    _Float16* __restrict__ hbuf) {      // [E,128]
  __shared__ __align__(16) _Float16 sA[16 * 32];
  int e0 = blockIdx.x * 16;
  int tid = threadIdx.x;
  for (int i = tid; i < 512; i += 128) {
    int m = i >> 5, k = i & 31;
    sA[i] = (k < 16) ? (_Float16)ea[(size_t)(e0 + m) * 16 + k] : (_Float16)0.f;
  }
  __syncthreads();
  int wave = tid >> 5, L = tid & 31;
  int m = L & 15, half = L >> 4, n = L & 15;
  v16h a = load_a_frag(&sA[m * 32], half);
  for (int t = wave; t < 8; t += 4) {
    v16h b = *(const v16h*)(w1f + ((size_t)t * 32 + L) * 16);
    v8f c = {};
    c = __builtin_amdgcn_wmma_f32_16x16x32_f16(false, a, false, b, (short)0,
                                               c, false, false);
    int col = t * 16 + n;
    float bb = b1[col];
#pragma unroll
    for (int v = 0; v < 8; ++v) {
      float o = c[v] + bb;
      o = o > 0.f ? o : 0.f;
      hbuf[(size_t)(e0 + v + 8 * half) * 128 + col] = (_Float16)o;
    }
  }
}

// ---------------------------------------------------------------------------
// Fused message GEMM, M-blocked by MB=5 subtiles (80 edges / block).
// acc[s] accumulates Z(s) @ w2' over C = (128*MI + MI)/32 WMMA chunks; each
// B fragment is loaded once and fed to 5 independent WMMAs.
// A rows: Z[m, kk] = h[m, kk/MI] * x_src[m, kk%MI]  (kk/MI==128 -> h:=1).
// ---------------------------------------------------------------------------
template <int MI, int MO>
__global__ __launch_bounds__(32 * (MO / 16)) void k_edge_msg(
    const float* __restrict__ x,        // [N, MI]
    const _Float16* __restrict__ hbuf,  // [E, 128]
    const _Float16* __restrict__ w2f,   // frags: C x T x 32 x 16
    const int* __restrict__ srcIdx,     // [E]
    const int* __restrict__ dstIdx,     // [E]
    float* __restrict__ agg) {          // [N, MO]
  constexpr int T = MO / 16;
  constexpr int NT = 32 * T;
  constexpr int MB = 5;                       // 80 edges per block
  constexpr int C = (128 * MI + MI) / 32;
  static_assert(NN_EDGES % (MB * 16) == 0, "edge tiling must be exact");
  __shared__ __align__(16) _Float16 sX[MB * 16 * MI];
  __shared__ __align__(16) _Float16 sH[MB * 16 * 128];
  __shared__ int sSrc[MB * 16], sDst[MB * 16];

  int e0 = blockIdx.x * (MB * 16);
  int tid = threadIdx.x;
  for (int i = tid; i < MB * 16; i += NT) {
    sSrc[i] = srcIdx[e0 + i];
    sDst[i] = dstIdx[e0 + i];
  }
  __syncthreads();
  // gather x rows (f32 -> f16 LDS)
  for (int i = tid; i < MB * 16 * MI; i += NT) {
    int r = i / MI, cc = i % MI;
    sX[i] = (_Float16)x[(size_t)sSrc[r] * MI + cc];
  }
  // h tiles are one contiguous 20 KB block: vector copy
  {
    const uint4* hs = (const uint4*)(hbuf + (size_t)e0 * 128);
    uint4* hd = (uint4*)sH;
    for (int i = tid; i < MB * 16 * 128 / 8; i += NT) hd[i] = hs[i];
  }
  __syncthreads();

  int wave = tid >> 5, L = tid & 31;
  int mrow = L & 15, half = L >> 4, n = L & 15;
  int t = wave;  // output column tile (one per wave)
  v8f zero = {};
  v8f acc[MB];
#pragma unroll
  for (int s = 0; s < MB; ++s) acc[s] = zero;

  for (int c = 0; c < C; ++c) {
    int kkb = c * 32;
    int k = kkb / MI;   // constant across the 32-chunk (MI is 32 or 64)
    int i0 = kkb % MI;
    v16h b = *(const v16h*)(w2f + (((size_t)c * T + t) * 32 + L) * 16);
    if (c + 1 < C)
      __builtin_prefetch(w2f + (((size_t)(c + 1) * T + t) * 32 + L) * 16, 0, 0);
#pragma unroll
    for (int s = 0; s < MB; ++s) {
      int row = s * 16 + mrow;
      _Float16 hm = (k < 128) ? sH[row * 128 + k] : (_Float16)1.0f;
      const _Float16* xb = &sX[row * MI + i0];
      v8h alo = (*(const v8h*)(xb + 8 * half)) * hm;        // v_pk_mul_f16 x4
      v8h ahi = (*(const v8h*)(xb + 16 + 8 * half)) * hm;
      v16h a = __builtin_shufflevector(alo, ahi, 0, 1, 2, 3, 4, 5, 6, 7, 8, 9,
                                       10, 11, 12, 13, 14, 15);
      acc[s] = __builtin_amdgcn_wmma_f32_16x16x32_f16(false, a, false, b,
                                                      (short)0, acc[s], false,
                                                      false);
    }
  }
  // scatter-add per-edge rows to destination nodes
#pragma unroll
  for (int s = 0; s < MB; ++s) {
#pragma unroll
    for (int v = 0; v < 8; ++v) {
      int m = s * 16 + v + 8 * half;
      atomicAdd(&agg[(size_t)sDst[m] * MO + t * 16 + n], acc[s][v]);
    }
  }
}

// ---------------------------------------------------------------------------
// x_new = relu(agg + x @ root + bias)   (root GEMM via WMMA, K = MI)
// ---------------------------------------------------------------------------
template <int MI, int MO>
__global__ __launch_bounds__(32 * (MO / 16)) void k_node_update(
    const float* __restrict__ x,        // [N, MI]
    const _Float16* __restrict__ rootf, // frags: (MI/32) x T x 32 x 16
    const float* __restrict__ bias,     // [MO]
    const float* __restrict__ agg,      // [N, MO]
    float* __restrict__ xout) {         // [N, MO]
  constexpr int T = MO / 16;
  constexpr int NT = 32 * T;
  constexpr int C = MI / 32;
  __shared__ __align__(16) _Float16 sX[16 * MI];
  int n0 = blockIdx.x * 16, tid = threadIdx.x;
  const float* xs = x + (size_t)n0 * MI;  // 16 rows are contiguous
  for (int i = tid; i < 16 * MI; i += NT) sX[i] = (_Float16)xs[i];
  __syncthreads();

  int wave = tid >> 5, L = tid & 31;
  int mrow = L & 15, half = L >> 4, n = L & 15;
  int t = wave;
  v8f acc = {};
  for (int c = 0; c < C; ++c) {
    v16h a = load_a_frag(&sX[mrow * MI + c * 32], half);
    v16h b = *(const v16h*)(rootf + (((size_t)c * T + t) * 32 + L) * 16);
    acc = __builtin_amdgcn_wmma_f32_16x16x32_f16(false, a, false, b, (short)0,
                                                 acc, false, false);
  }
  int col = t * 16 + n;
  float bb = bias[col];
#pragma unroll
  for (int v = 0; v < 8; ++v) {
    int node = n0 + v + 8 * half;
    float o = acc[v] + agg[(size_t)node * MO + col] + bb;
    xout[(size_t)node * MO + col] = o > 0.f ? o : 0.f;
  }
}

// ---------------------------------------------------------------------------
// Graph pooling: g[batch[n]] += x[n]  (f32 atomics)
// ---------------------------------------------------------------------------
__global__ void k_pool(const float* __restrict__ x,
                       const int* __restrict__ batch,
                       float* __restrict__ g) {
  int total = NN_NODES * 64;
  for (int idx = blockIdx.x * blockDim.x + threadIdx.x; idx < total;
       idx += gridDim.x * blockDim.x) {
    int node = idx >> 6, f = idx & 63;
    atomicAdd(&g[(size_t)batch[node] * 64 + f], x[idx]);
  }
}

// ---------------------------------------------------------------------------
// Output head: 64 -> 32 -> 16 -> 1 (tiny; one thread per graph)
// ---------------------------------------------------------------------------
__global__ __launch_bounds__(64) void k_head(
    const float* __restrict__ g,
    const float* __restrict__ w0, const float* __restrict__ b0,
    const float* __restrict__ w1, const float* __restrict__ b1,
    const float* __restrict__ w2, const float* __restrict__ b2,
    float* __restrict__ out) {
  int gi = blockIdx.x * blockDim.x + threadIdx.x;
  if (gi >= NN_GRAPHS) return;
  float v0[64], v1[32], v2[16];
#pragma unroll
  for (int i = 0; i < 64; ++i) v0[i] = g[(size_t)gi * 64 + i];
#pragma unroll
  for (int o = 0; o < 32; ++o) {
    float s = b0[o];
    for (int i = 0; i < 64; ++i) s += v0[i] * w0[i * 32 + o];
    v1[o] = fmaxf(s, 0.f);
  }
#pragma unroll
  for (int o = 0; o < 16; ++o) {
    float s = b1[o];
    for (int i = 0; i < 32; ++i) s += v1[i] * w1[i * 16 + o];
    v2[o] = fmaxf(s, 0.f);
  }
  float s = b2[0];
#pragma unroll
  for (int i = 0; i < 16; ++i) s += v2[i] * w2[i];
  out[gi] = s;
}

// ---------------------------------------------------------------------------
extern "C" void kernel_launch(void* const* d_in, const int* in_sizes, int n_in,
                              void* d_out, int out_size, void* d_ws,
                              size_t ws_size, hipStream_t stream) {
  (void)in_sizes; (void)n_in; (void)out_size; (void)ws_size;

  const float* x_in = (const float*)d_in[0];
  const float* ea   = (const float*)d_in[1];
  const int* eidx   = (const int*)d_in[26];
  const int* src    = eidx;              // edge_index[0]
  const int* dst    = eidx + NN_EDGES;   // edge_index[1]
  const int* batch  = (const int*)d_in[27];
  float* out        = (float*)d_out;

  const int MIs[3] = {64, 32, 64};
  const int MOs[3] = {32, 64, 64};

  // workspace carve-out (~40 MB), 256B-aligned slices
  char* ws = (char*)d_ws;
  size_t off = 0;
  auto take = [&](size_t bytes) -> char* {
    off = (off + 255) & ~(size_t)255;
    char* p = ws + off;
    off += bytes;
    return p;
  };
  _Float16* hbuf = (_Float16*)take((size_t)NN_EDGES * 128 * 2);
  _Float16 *w1f[3], *w2f[3], *rootf[3];
  int w2C[3], rC[3];
  for (int l = 0; l < 3; ++l) {
    w1f[l] = (_Float16*)take((size_t)8 * 512 * 2);               // C=1, T=8
    w2C[l] = (128 * MIs[l] + MIs[l]) / 32;
    w2f[l] = (_Float16*)take((size_t)w2C[l] * (MOs[l] / 16) * 512 * 2);
    rC[l] = MIs[l] / 32;
    rootf[l] = (_Float16*)take((size_t)rC[l] * (MOs[l] / 16) * 512 * 2);
  }
  float* agg = (float*)take((size_t)NN_NODES * 64 * 4);
  float* xA  = (float*)take((size_t)NN_NODES * 64 * 4);
  float* xB  = (float*)take((size_t)NN_NODES * 64 * 4);
  float* g   = (float*)take((size_t)NN_GRAPHS * 64 * 4);

  // --- weight prep: f32 -> swizzled f16 B fragments (recomputed every call,
  //     deterministic; tiny cost, fully L2-resident thereafter) ---
  for (int l = 0; l < 3; ++l) {
    const float* w1   = (const float*)d_in[2 + l * 6 + 0];
    const float* w2   = (const float*)d_in[2 + l * 6 + 2];
    const float* b2   = (const float*)d_in[2 + l * 6 + 3];
    const float* root = (const float*)d_in[2 + l * 6 + 4];
    int tot1 = 8 * 512;
    k_swizzleB<<<(tot1 + 255) / 256, 256, 0, stream>>>(
        w1, nullptr, 16, 0, 128, w1f[l], tot1);
    int tot2 = w2C[l] * (MOs[l] / 16) * 512;
    k_swizzleB<<<(tot2 + 255) / 256, 256, 0, stream>>>(
        w2, b2, 128 * MIs[l], MIs[l], MOs[l], w2f[l], tot2);
    int totr = rC[l] * (MOs[l] / 16) * 512;
    k_swizzleB<<<(totr + 255) / 256, 256, 0, stream>>>(
        root, nullptr, MIs[l], 0, MOs[l], rootf[l], totr);
  }

  // --- three NNConv layers ---
  const float* xcur = x_in;
  float* xnext = xA;
  for (int l = 0; l < 3; ++l) {
    const float* b1   = (const float*)d_in[2 + l * 6 + 1];
    const float* bias = (const float*)d_in[2 + l * 6 + 5];
    k_edge_h<<<NN_EDGES / 16, 128, 0, stream>>>(ea, w1f[l], b1, hbuf);
    (void)hipMemsetAsync(agg, 0, (size_t)NN_NODES * MOs[l] * 4, stream);
    if (l == 0) {
      k_edge_msg<64, 32><<<NN_EDGES / 80, 64, 0, stream>>>(
          xcur, hbuf, w2f[l], src, dst, agg);
      k_node_update<64, 32><<<NN_NODES / 16, 64, 0, stream>>>(
          xcur, rootf[l], bias, agg, xnext);
    } else if (l == 1) {
      k_edge_msg<32, 64><<<NN_EDGES / 80, 128, 0, stream>>>(
          xcur, hbuf, w2f[l], src, dst, agg);
      k_node_update<32, 64><<<NN_NODES / 16, 128, 0, stream>>>(
          xcur, rootf[l], bias, agg, xnext);
    } else {
      k_edge_msg<64, 64><<<NN_EDGES / 80, 128, 0, stream>>>(
          xcur, hbuf, w2f[l], src, dst, agg);
      k_node_update<64, 64><<<NN_NODES / 16, 128, 0, stream>>>(
          xcur, rootf[l], bias, agg, xnext);
    }
    xcur = xnext;
    xnext = (xnext == xA) ? xB : xA;
  }

  // --- pooling + head ---
  (void)hipMemsetAsync(g, 0, (size_t)NN_GRAPHS * 64 * 4, stream);
  k_pool<<<7500, 256, 0, stream>>>(xcur, batch, g);
  k_head<<<(NN_GRAPHS + 63) / 64, 64, 0, stream>>>(
      g, (const float*)d_in[20], (const float*)d_in[21],
      (const float*)d_in[22], (const float*)d_in[23],
      (const float*)d_in[24], (const float*)d_in[25], out);
}
